// RNN_74139725464046
// MI455X (gfx1250) — compile-verified, matrix operand order checked by reading, not run on hardware
//
#include <hip/hip_runtime.h>
#include <cmath>

// LSTM: T=2048, B=256, I=64, H=128, O=64. Fully fused persistent kernel.
// Grid = 16 workgroups (one per 16-row batch tile), 512 threads = 16 wave32.
// Weights register-resident as f16 WMMA B-fragments for all 2048 steps;
// biases folded into WMMA accumulator init (C/D column == gate index);
// A-fragments double-buffered; activations branch-free on native v_exp/v_rcp.

#define T_STEPS 2048
#define BATCH   256
#define IND     64
#define HID     128
#define OUTD    64
#define G4      (4 * HID)      // 512 gate rows
#define BT      16             // batch rows per workgroup
#define NWAVES  16
#define NTHREADS (NWAVES * 32)
#define KDIM    (IND + HID)    // 192 = fused K for [x | h]
#define ASTR    200            // padded f16 K-stride for A operand (16B-aligned rows)
#define GSTR    520            // padded f32 stride for gates buffer

typedef __attribute__((ext_vector_type(16))) _Float16 v16h;
typedef __attribute__((ext_vector_type(8)))  _Float16 v8h;
typedef __attribute__((ext_vector_type(4)))  _Float16 v4h;
typedef __attribute__((ext_vector_type(2)))  _Float16 v2h;
typedef __attribute__((ext_vector_type(8)))  float    v8f;

union U16h { v16h v; v8h h[2]; };
union UF4  { float4 v; float a[4]; };

// convert 8 consecutive f32 -> v8h (used once at startup to build register B-frags)
static __device__ inline v8h cvt8(const float* __restrict__ p) {
  v8h r;
#pragma unroll
  for (int e = 0; e < 8; ++e) r[e] = (_Float16)p[e];
  return r;
}

// Load one 16x32 f16 A-fragment (wave32 WMMA layout) from an LDS row-major tile.
// Lane (m = lane&15, half = lane>>4): elems 0..7  = K = kbase + half*8 + {0..7}
//                                     elems 8..15 = K = kbase + 16 + half*8 + {0..7}
static __device__ inline v16h lds_a_frag(const _Float16* __restrict__ row,
                                         int kbase, int half) {
  U16h u;
  u.h[0] = *(const v8h*)(row + kbase + half * 8);
  u.h[1] = *(const v8h*)(row + kbase + 16 + half * 8);
  return u.v;
}

// Branch-free fast activations: native v_exp_f32 / v_rcp_f32, no OCML branches.
static __device__ inline float fast_sigmoid(float x) {
  return __builtin_amdgcn_rcpf(1.0f + __expf(-x));   // denom in (1,inf): rcp safe
}
static __device__ inline float fast_tanh(float x) {
  float ax = __builtin_fabsf(x);
  float e  = __expf(-2.0f * ax);                      // in (0,1]
  float r  = (1.0f - e) * __builtin_amdgcn_rcpf(1.0f + e);
  return __builtin_copysignf(r, x);
}

__global__ __launch_bounds__(NTHREADS)
void lstm_fused_kernel(const float* __restrict__ x,     // [T,B,I]
                       const float* __restrict__ Wih,   // [4H,I]
                       const float* __restrict__ Whh,   // [4H,H]
                       const float* __restrict__ bih,   // [4H]
                       const float* __restrict__ bhh,   // [4H]
                       const float* __restrict__ Wlin,  // [O,H]
                       const float* __restrict__ blin,  // [O]
                       float* __restrict__ out)         // [B,T,O]
{
  __shared__ _Float16 Ash[BT * ASTR];   // per-row fused A operand: [x(0..63) | h(64..191)] f16
  __shared__ float    gsh[BT * GSTR];   // gates [16][512] f32 (bias already included)
  __shared__ float    csh[BT * HID];    // cell state f32

  const int tid  = threadIdx.x;
  const int lane = tid & 31;            // wave32
  const int wave = tid >> 5;
  const int n    = lane & 15;
  const int half = lane >> 4;
  const int b0   = blockIdx.x * BT;

  // ---- one-time init: zero h and c ----
  for (int i = tid; i < BT * HID; i += NTHREADS)  csh[i] = 0.0f;
  for (int i = tid; i < BT * ASTR; i += NTHREADS) Ash[i] = (_Float16)0.0f;

  // ---- one-time: per-lane biases (C/D column == gate row g, fixed per lane) ----
  const int g0tile = (wave * 2 + 0) * 16 + n;
  const int g1tile = (wave * 2 + 1) * 16 + n;
  const float bias0 = bih[g0tile] + bhh[g0tile];
  const float bias1 = bih[g1tile] + bhh[g1tile];
  const float biasl = blin[(wave & 3) * 16 + n];

  // ---- one-time: register-resident B fragments (f32 -> f16) ----
  // Gate GEMM B operand: B[k][n] = W[g0+n][k], K chunks c=0..5 over [x|h]:
  //   c<2  -> W_ih (k in [0,64)),  c>=2 -> W_hh (k-64 in [0,128))
  v16h bfrag[2][6];
#pragma unroll
  for (int j = 0; j < 2; ++j) {
    const int g = (wave * 2 + j) * 16 + n;
#pragma unroll
    for (int c = 0; c < 6; ++c) {
      const int ka = c * 32 + half * 8;
      U16h u;
      if (c < 2) {
        const float* row = Wih + g * IND;
        u.h[0] = cvt8(row + ka);
        u.h[1] = cvt8(row + ka + 16);
      } else {
        const float* row = Whh + g * HID;
        u.h[0] = cvt8(row + (ka - 64));
        u.h[1] = cvt8(row + (ka - 64) + 16);
      }
      bfrag[j][c] = u.v;
    }
  }
  // Output-projection B operand: B[k][n] = W_lin[o0+n][k]; used by waves 0..3.
  v16h blf[4];
#pragma unroll
  for (int c = 0; c < 4; ++c) {
    const int o = (wave & 3) * 16 + n;
    const float* row = Wlin + o * HID;
    const int ka = c * 32 + half * 8;
    U16h u;
    u.h[0] = cvt8(row + ka);
    u.h[1] = cvt8(row + ka + 16);
    blf[c] = u.v;
  }

  const int xrow = tid >> 5;            // 0..15
  const int xcol = (tid & 31) * 2;      // 0..62 (2 elems/thread)
  const int erow = tid >> 5;            // elementwise phase
  const int ehid = (tid & 31) * 4;      // 0..124 (4 elems/thread)
  const _Float16* arow = Ash + n * ASTR;  // A-frag row for this lane (m == n)

  for (int t = 0; t < T_STEPS; ++t) {
    // ---- phase 0: stage x_t (f32 -> f16) into A region; prefetch x_{t+1} ----
    {
      const float* xp = x + ((size_t)t * BATCH + (b0 + xrow)) * IND + xcol;
      float2 xv = *(const float2*)xp;
      v2h hx; hx[0] = (_Float16)xv.x; hx[1] = (_Float16)xv.y;
      *(v2h*)(Ash + xrow * ASTR + xcol) = hx;
      if (t + 1 < T_STEPS)
        __builtin_prefetch(xp + BATCH * IND, 0, 1);   // -> global_prefetch_b8
    }
    __syncthreads();

    // ---- phase A: gates[16,512] = [x_t | h_{t-1}] x [W_ih | W_hh]^T + b  via WMMA ----
    {
      // Double-buffered A fragments: issue loads for chunk c+1 before the
      // WMMAs of chunk c so ds latency overlaps matrix math (partial dscnt wait).
      v16h a_cur = lds_a_frag(arow, 0, half);
      v8f acc0, acc1;  // init with bias (column == gate row, per-lane scalar)
#pragma unroll
      for (int r = 0; r < 8; ++r) { acc0[r] = bias0; acc1[r] = bias1; }

#pragma unroll
      for (int c = 0; c < 6; ++c) {
        v16h a_nxt;
        if (c < 5) a_nxt = lds_a_frag(arow, (c + 1) * 32, half);
        acc0 = __builtin_amdgcn_wmma_f32_16x16x32_f16(
            false, a_cur, false, bfrag[0][c], (short)0, acc0, false, false);
        acc1 = __builtin_amdgcn_wmma_f32_16x16x32_f16(
            false, a_cur, false, bfrag[1][c], (short)0, acc1, false, false);
        a_cur = a_nxt;
      }
      // D layout: VGPR r holds row m = r + 8*half, col n
#pragma unroll
      for (int r = 0; r < 8; ++r) {
        const int m = r + 8 * half;
        gsh[m * GSTR + (wave * 2 + 0) * 16 + n] = acc0[r];
        gsh[m * GSTR + (wave * 2 + 1) * 16 + n] = acc1[r];
      }
    }
    __syncthreads();

    // ---- phase B: elementwise cell update; h_t written back as f16 A operand ----
    {
      UF4 iu, fu, gu, ou, cu;
      iu.v = *(const float4*)(gsh + erow * GSTR + 0 * HID + ehid);
      fu.v = *(const float4*)(gsh + erow * GSTR + 1 * HID + ehid);
      gu.v = *(const float4*)(gsh + erow * GSTR + 2 * HID + ehid);
      ou.v = *(const float4*)(gsh + erow * GSTR + 3 * HID + ehid);
      cu.v = *(const float4*)(csh + erow * HID + ehid);
      v4h hh;
#pragma unroll
      for (int e = 0; e < 4; ++e) {
        float ig = fast_sigmoid(iu.a[e]);
        float fg = fast_sigmoid(fu.a[e]);
        float gg = fast_tanh(gu.a[e]);
        float og = fast_sigmoid(ou.a[e]);
        float cn = fg * cu.a[e] + ig * gg;
        cu.a[e] = cn;
        hh[e] = (_Float16)(og * fast_tanh(cn));
      }
      *(float4*)(csh + erow * HID + ehid) = cu.v;
      *(v4h*)(Ash + erow * ASTR + IND + ehid) = hh;
    }
    __syncthreads();

    // ---- phase C: out[b, t, :] = h_t @ W_lin^T + b_lin (waves 0..3, EXEC full) ----
    if (wave < 4) {
      v16h a_cur = lds_a_frag(arow, IND, half);
      v8f acc;  // init with b_lin (column-indexed, per-lane scalar)
#pragma unroll
      for (int r = 0; r < 8; ++r) acc[r] = biasl;

#pragma unroll
      for (int c = 0; c < 4; ++c) {
        v16h a_nxt;
        if (c < 3) a_nxt = lds_a_frag(arow, IND + (c + 1) * 32, half);
        acc = __builtin_amdgcn_wmma_f32_16x16x32_f16(
            false, a_cur, false, blf[c], (short)0, acc, false, false);
        a_cur = a_nxt;
      }
      const int col = wave * 16 + n;
#pragma unroll
      for (int r = 0; r < 8; ++r) {
        const int m = r + 8 * half;
        __builtin_nontemporal_store(
            acc[r], &out[((size_t)(b0 + m) * T_STEPS + t) * OUTD + col]);
      }
    }
    // top-of-loop barrier (after next x stage) orders phase C reads vs next phase B writes
  }
}

extern "C" void kernel_launch(void* const* d_in, const int* in_sizes, int n_in,
                              void* d_out, int out_size, void* d_ws, size_t ws_size,
                              hipStream_t stream) {
  (void)in_sizes; (void)n_in; (void)d_ws; (void)ws_size; (void)out_size;
  const float* x    = (const float*)d_in[0];
  const float* Wih  = (const float*)d_in[1];
  const float* Whh  = (const float*)d_in[2];
  const float* bih  = (const float*)d_in[3];
  const float* bhh  = (const float*)d_in[4];
  const float* Wlin = (const float*)d_in[5];
  const float* blin = (const float*)d_in[6];
  float* out = (float*)d_out;

  lstm_fused_kernel<<<dim3(BATCH / BT), dim3(NTHREADS), 0, stream>>>(
      x, Wih, Whh, bih, bhh, Wlin, blin, out);
}
